// MambaLite_83339545411727
// MI455X (gfx1250) — matrix operand order repaired and checked by reading, not compiled, not run on hardware
//
#include <hip/hip_runtime.h>
#include <hip/hip_bf16.h>

// ---------------------------------------------------------------------------
// MambaLite block for MI455X (gfx1250, wave32, WMMA + TDM + async global->LDS)
//   x -> LN -> GEMM1(in_w) -> split u,v,g -> dwconv3+silu(u) -> SSM scan
//     -> y*sigmoid(g)+silu(v) -> GEMM2(out_w) + bias + residual
// GEMMs: v_wmma_f32_16x16x32_f16.
//   A tiles: per-thread global_load_async_to_lds_b128 (ASYNCcnt)
//   B tiles: Tensor Data Mover tensor_load_to_lds, one DMA per block/stage,
//            LDS pad via D# (pad_interval=64dw, pad_amount=4dw) (TENSORcnt)
// Double-buffered with explicit s_wait_asynccnt / s_wait_tensorcnt.
// ---------------------------------------------------------------------------

#define D_MODEL 1024
#define L_SEQ   1024
#define INNER   2048
#define N3      (3 * INNER)   // 6144
#define D_STATE 64
#define LN_EPS  1e-5f

// GEMM tiling
#define BM 64
#define BN 128
#define BK 32
#define LDA_S 40    // padded halfs per A row in LDS (80B, bank-friendly)
#define LDB_S 136   // padded halfs per B row in LDS (272B = 64+4 dwords/row)

typedef __attribute__((ext_vector_type(16))) _Float16 v16h;
typedef __attribute__((ext_vector_type(8)))  _Float16 v8h;
typedef __attribute__((ext_vector_type(8)))  float    v8f;
typedef __attribute__((ext_vector_type(4)))  int      v4i;
typedef __attribute__((ext_vector_type(4)))  unsigned u32x4;
typedef __attribute__((ext_vector_type(8)))  int      i32x8;
typedef __attribute__((ext_vector_type(4)))  int      i32x4;

#define AS1 __attribute__((address_space(1)))
#define AS3 __attribute__((address_space(3)))

// ------------------------------------------------- async global->LDS helpers
__device__ __forceinline__ void async_copy_b128(const _Float16* g, _Float16* l) {
#if __has_builtin(__builtin_amdgcn_global_load_async_to_lds_b128)
    __builtin_amdgcn_global_load_async_to_lds_b128(
        (AS1 v4i*)g, (AS3 v4i*)l, 0, 0);
#else
    const unsigned loff = (unsigned)(unsigned long long)(AS3 void*)l;
    asm volatile("global_load_async_to_lds_b128 %0, %1, off"
                 :: "v"(loff), "v"(g) : "memory");
#endif
}

__device__ __forceinline__ void wait_async_0() {
#if __has_builtin(__builtin_amdgcn_s_wait_asynccnt)
    __builtin_amdgcn_s_wait_asynccnt(0);
#else
    asm volatile("s_wait_asynccnt 0x0" ::: "memory");
#endif
}

__device__ __forceinline__ void wait_async_1() {
#if __has_builtin(__builtin_amdgcn_s_wait_asynccnt)
    __builtin_amdgcn_s_wait_asynccnt(1);
#else
    asm volatile("s_wait_asynccnt 0x1" ::: "memory");
#endif
}

__device__ __forceinline__ void wait_async_3() {
#if __has_builtin(__builtin_amdgcn_s_wait_asynccnt)
    __builtin_amdgcn_s_wait_asynccnt(3);
#else
    asm volatile("s_wait_asynccnt 0x3" ::: "memory");
#endif
}

// ---------------------------------------------------------- TDM B-tile load
#if __has_builtin(__builtin_amdgcn_tensor_load_to_lds)
#define HAVE_TDM 1

__device__ __forceinline__ void wait_tensor_0() {
#if __has_builtin(__builtin_amdgcn_s_wait_tensorcnt)
    __builtin_amdgcn_s_wait_tensorcnt(0);
#else
    asm volatile("s_wait_tensorcnt 0x0" ::: "memory");
#endif
}
__device__ __forceinline__ void wait_tensor_1() {
#if __has_builtin(__builtin_amdgcn_s_wait_tensorcnt)
    __builtin_amdgcn_s_wait_tensorcnt(1);
#else
    asm volatile("s_wait_tensorcnt 0x1" ::: "memory");
#endif
}

// One 2D DMA: tile = 32 rows x 64 dwords (BK x BN halfs) from row-major B
// (row stride ldb_h halfs) into LDS with 4 dwords of pad after every 64
// dwords -> padded LDS row stride = 272B = LDB_S halfs.
__device__ __forceinline__ void tdm_load_b_tile(const _Float16* gsrc,
                                                _Float16* ldst, int ldb_h) {
    const unsigned long long ga = (unsigned long long)gsrc;
    const unsigned la = (unsigned)(unsigned long long)(AS3 void*)ldst;

    u32x4 g0;
    g0[0] = 1u;                                            // count=1, user mode
    g0[1] = la;                                            // lds_addr (bytes)
    g0[2] = (unsigned)(ga & 0xFFFFFFFFu);                  // global_addr[31:0]
    g0[3] = (unsigned)((ga >> 32) & 0x01FFFFFFu)           // global_addr[56:32]
          | (2u << 30);                                    // type=2 (image)

    i32x8 g1;
    g1[0] = (2 << 16)        // data_size = 4B elements
          | (1 << 20)        // pad_enable
          | (5 << 22)        // pad_interval: every 64 dwords
          | (3 << 25);       // pad_amount: 4 dwords
    g1[1] = (BN / 2) << 16;                 // tensor_dim0 = 64 (bits 79:48 lo)
    g1[2] = (BK) << 16;                     // dim0 hi=0 | tensor_dim1 = 32
    g1[3] = (BN / 2) << 16;                 // dim1 hi=0 | tile_dim0 = 64
    g1[4] = BK;                             // tile_dim1 = 32, tile_dim2 = 0
    g1[5] = ldb_h >> 1;                     // tensor_dim0_stride (dwords) lo32
    g1[6] = 0;                              // stride hi | dim1_stride lo
    g1[7] = 0;

    const i32x4 z4 = {0, 0, 0, 0};              // groups 2/3 unused (2-D)
    const i32x8 z8 = {0, 0, 0, 0, 0, 0, 0, 0};  // extra group (6-arg form)
    __builtin_amdgcn_tensor_load_to_lds(g0, g1, z4, z4, z8, 0);
}
#else
#define HAVE_TDM 0
#endif

// ---------------------------------------------------------------- utilities
__global__ void f32_to_f16_kernel(const float* __restrict__ src,
                                  _Float16* __restrict__ dst, int n) {
    int i = blockIdx.x * blockDim.x + threadIdx.x;
    if (i < n) dst[i] = (_Float16)src[i];
}

// ------------------------------------------------------------- layernorm
__global__ void layernorm_f16_kernel(const float* __restrict__ x,
                                     const float* __restrict__ gamma,
                                     const float* __restrict__ beta,
                                     _Float16* __restrict__ h16) {
    __shared__ float red[256];
    const int row = blockIdx.x;
    const int tid = threadIdx.x;
    const float4 v = reinterpret_cast<const float4*>(x + row * D_MODEL)[tid];

    red[tid] = v.x + v.y + v.z + v.w;
    __syncthreads();
    for (int off = 128; off > 0; off >>= 1) {
        if (tid < off) red[tid] += red[tid + off];
        __syncthreads();
    }
    const float mu = red[0] * (1.0f / D_MODEL);
    __syncthreads();

    const float d0 = v.x - mu, d1 = v.y - mu, d2 = v.z - mu, d3 = v.w - mu;
    red[tid] = d0 * d0 + d1 * d1 + d2 * d2 + d3 * d3;
    __syncthreads();
    for (int off = 128; off > 0; off >>= 1) {
        if (tid < off) red[tid] += red[tid + off];
        __syncthreads();
    }
    const float rs = rsqrtf(red[0] * (1.0f / D_MODEL) + LN_EPS);

    const int i = tid * 4;
    _Float16* out = h16 + row * D_MODEL + i;
    out[0] = (_Float16)(d0 * rs * gamma[i + 0] + beta[i + 0]);
    out[1] = (_Float16)(d1 * rs * gamma[i + 1] + beta[i + 1]);
    out[2] = (_Float16)(d2 * rs * gamma[i + 2] + beta[i + 2]);
    out[3] = (_Float16)(d3 * rs * gamma[i + 3] + beta[i + 3]);
}

// --------------------------------------------------------------- WMMA GEMM
// C(MxN,f32) = A(MxK,f16) * B(KxN,f16) + bias [+ residual]
// Block: 256 threads = 8 waves; block tile 64x128, K-step 32.
// Waves arranged 2(M) x 4(N); each wave: 32x32 = 2x2 WMMA tiles.
// Fragment layouts per CDNA5 ISA 7.12.2:
//   A 16x32 f16 : lanes 0-15 row M=lane, K in {0..7,16..23};
//                 lanes 16-31 row M=lane-16, K in {8..15,24..31}
//   B 32x16 f16 : lane&15 <-> N col; 16 halfs = K, same order as A
//   C/D 16x16 f32: VGPR i -> M = i + 8*(lane>=16), N = lane&15
__global__ void wmma_gemm_f16_kernel(const _Float16* __restrict__ A,
                                     const _Float16* __restrict__ B,
                                     const float* __restrict__ bias,
                                     const float* __restrict__ residual,
                                     float* __restrict__ C,
                                     int K, int lda, int ldb, int ldc) {
    __shared__ _Float16 sA[2][BM * LDA_S];
    __shared__ _Float16 sB[2][BK * LDB_S];

    const int tid  = threadIdx.x;
    const int n0   = blockIdx.x * BN;
    const int m0   = blockIdx.y * BM;
    const int w    = tid >> 5;          // wave 0..7
    const int lane = tid & 31;
    const int r    = lane & 15;
    const int h    = lane >> 4;
    const int mw   = (w >> 2) * 32;     // 0 or 32
    const int nw   = (w & 3) * 32;      // 0,32,64,96

    // per-thread async-copy coordinates (A tile)
    const int arow = tid >> 2;          // 0..63
    const int acol = (tid & 3) * 8;     // halfs
#if !HAVE_TDM
    const int brow = tid >> 3;          // 0..31
    const int bc0  = (tid & 7) * 16;    // halfs (two 8-half chunks)
#endif

    const int nsteps = K / BK;

    // ---- stage 0 issue
    async_copy_b128(A + (size_t)(m0 + arow) * lda + acol,
                    &sA[0][arow * LDA_S + acol]);
#if HAVE_TDM
    if (w == 0) tdm_load_b_tile(B + n0, &sB[0][0], ldb);
#else
    async_copy_b128(B + (size_t)brow * ldb + n0 + bc0,
                    &sB[0][brow * LDB_S + bc0]);
    async_copy_b128(B + (size_t)brow * ldb + n0 + bc0 + 8,
                    &sB[0][brow * LDB_S + bc0 + 8]);
#endif

    v8f acc[2][2] = {};

    for (int ks = 0; ks < nsteps; ++ks) {
        const int buf = ks & 1;
        if (ks + 1 < nsteps) {
            const int nb = buf ^ 1;
            const int k1 = (ks + 1) * BK;
            async_copy_b128(A + (size_t)(m0 + arow) * lda + k1 + acol,
                            &sA[nb][arow * LDA_S + acol]);
#if HAVE_TDM
            if (w == 0) tdm_load_b_tile(B + (size_t)k1 * ldb + n0, &sB[nb][0], ldb);
            wait_async_1();                 // A stage ks landed; ks+1 in flight
            if (w == 0) wait_tensor_1();    // B stage ks landed; ks+1 in flight
#else
            async_copy_b128(B + (size_t)(k1 + brow) * ldb + n0 + bc0,
                            &sB[nb][brow * LDB_S + bc0]);
            async_copy_b128(B + (size_t)(k1 + brow) * ldb + n0 + bc0 + 8,
                            &sB[nb][brow * LDB_S + bc0 + 8]);
            wait_async_3();
#endif
            if (ks + 2 < nsteps)  // warm L2 for the stage after next
                __builtin_prefetch(B + (size_t)((ks + 2) * BK + (lane & 31)) * ldb + n0, 0, 3);
        } else {
            wait_async_0();
#if HAVE_TDM
            if (w == 0) wait_tensor_0();
#endif
        }
        __syncthreads();

        // A fragments (2 row tiles), from LDS
        v16h afrag[2];
#pragma unroll
        for (int mi = 0; mi < 2; ++mi) {
            const _Float16* ar = &sA[buf][(mw + mi * 16 + r) * LDA_S + 8 * h];
            const v8h lo = *reinterpret_cast<const v8h*>(ar);
            const v8h hi = *reinterpret_cast<const v8h*>(ar + 16);
            v16h a;
#pragma unroll
            for (int j = 0; j < 8; ++j) { a[j] = lo[j]; a[j + 8] = hi[j]; }
            afrag[mi] = a;
        }
        // B fragments (2 col tiles), lane <-> N col; K in {8h..8h+7, 16+8h..16+8h+7}
        v16h bfrag[2];
#pragma unroll
        for (int ni = 0; ni < 2; ++ni) {
            const _Float16* br = &sB[buf][(8 * h) * LDB_S + nw + ni * 16 + r];
            v16h b;
#pragma unroll
            for (int j = 0; j < 8; ++j) { b[j] = br[j * LDB_S]; b[j + 8] = br[(16 + j) * LDB_S]; }
            bfrag[ni] = b;
        }

#pragma unroll
        for (int mi = 0; mi < 2; ++mi)
#pragma unroll
            for (int ni = 0; ni < 2; ++ni)
                acc[mi][ni] = __builtin_amdgcn_wmma_f32_16x16x32_f16(
                    false, afrag[mi], false, bfrag[ni],
                    (short)0, acc[mi][ni], false, false);

        __syncthreads();
    }

    // epilogue: bias (+ residual) and store
#pragma unroll
    for (int mi = 0; mi < 2; ++mi) {
#pragma unroll
        for (int ni = 0; ni < 2; ++ni) {
            const int col  = n0 + nw + ni * 16 + r;
            const float bc = bias[col];
#pragma unroll
            for (int i = 0; i < 8; ++i) {
                const int row = m0 + mw + mi * 16 + i + 8 * h;
                float v = acc[mi][ni][i] + bc;
                if (residual) v += residual[(size_t)row * ldc + col];
                C[(size_t)row * ldc + col] = v;
            }
        }
    }
}

// ------------------------------------------------- depthwise conv3 + SiLU
__global__ void conv_silu_kernel(const float* __restrict__ uvg,
                                 const float* __restrict__ cw,
                                 const float* __restrict__ cb,
                                 float* __restrict__ ucv) {
    const int idx = blockIdx.x * blockDim.x + threadIdx.x;   // L*INNER
    if (idx >= L_SEQ * INNER) return;
    const int c = idx & (INNER - 1);
    const int l = idx >> 11;                                  // /INNER
    float acc = cb[c];
#pragma unroll
    for (int k = 0; k < 3; ++k) {
        const int ls = l + k - 1;
        if (ls >= 0 && ls < L_SEQ) acc += uvg[(size_t)ls * N3 + c] * cw[c * 3 + k];
    }
    ucv[idx] = acc / (1.0f + __expf(-acc));                   // silu
}

// -------------------------------------------------------------- SSM scan
__global__ void ssm_scan_kernel(const float* __restrict__ ucv,
                                const float* __restrict__ uvg,
                                const float* __restrict__ A_log,
                                const float* __restrict__ Bp,
                                const float* __restrict__ Cp,
                                const float* __restrict__ Dp,
                                _Float16* __restrict__ y16) {
    const int t    = blockIdx.x * blockDim.x + threadIdx.x;   // 0 .. 4*INNER-1
    const int c    = t >> 2;
    const int part = t & 3;
    const int sb   = part * 16;

    float eA[16], Bv[16], Cv[16], s[16];
#pragma unroll
    for (int i = 0; i < 16; ++i) {
        const int j = c * D_STATE + sb + i;
        eA[i] = __expf(-__expf(A_log[j]));
        Bv[i] = Bp[j];
        Cv[i] = Cp[j];
        s[i]  = 0.0f;
    }
    const float D = Dp[c];

    for (int l = 0; l < L_SEQ; ++l) {
        const float u = ucv[(size_t)l * INNER + c];
        float acc = 0.0f;
#pragma unroll
        for (int i = 0; i < 16; ++i) {
            s[i] = fmaf(s[i], eA[i], u * Bv[i]);
            acc  = fmaf(s[i], Cv[i], acc);
        }
        acc += __shfl_xor(acc, 1, 32);
        acc += __shfl_xor(acc, 2, 32);
        if (part == 0) {
            const float yt = fmaf(u, D, acc);
            const float g  = uvg[(size_t)l * N3 + 2 * INNER + c];
            const float v  = uvg[(size_t)l * N3 + INNER + c];
            const float sg = 1.0f / (1.0f + __expf(-g));
            const float sv = v / (1.0f + __expf(-v));
            y16[(size_t)l * INNER + c] = (_Float16)(yt * sg + sv);
        }
    }
}

// ---------------------------------------------------------------- launcher
extern "C" void kernel_launch(void* const* d_in, const int* in_sizes, int n_in,
                              void* d_out, int out_size, void* d_ws, size_t ws_size,
                              hipStream_t stream) {
    (void)in_sizes; (void)n_in; (void)out_size; (void)ws_size;

    const float* x      = (const float*)d_in[0];
    const float* gamma  = (const float*)d_in[1];
    const float* beta   = (const float*)d_in[2];
    const float* in_w   = (const float*)d_in[3];
    const float* in_b   = (const float*)d_in[4];
    const float* conv_w = (const float*)d_in[5];
    const float* conv_b = (const float*)d_in[6];
    const float* A_log  = (const float*)d_in[7];
    const float* B_p    = (const float*)d_in[8];
    const float* C_p    = (const float*)d_in[9];
    const float* D_p    = (const float*)d_in[10];
    const float* out_w  = (const float*)d_in[11];
    const float* out_b  = (const float*)d_in[12];
    float* out = (float*)d_out;

    char* ws = (char*)d_ws;
    _Float16* h16   = (_Float16*)(ws + 0);               //  2 MB
    _Float16* w1_16 = (_Float16*)(ws + (size_t)2097152); // 12 MB
    _Float16* w2_16 = (_Float16*)(ws + (size_t)14680064);//  4 MB
    float*    uvg   = (float*)   (ws + (size_t)18874368);// 25 MB
    float*    ucv   = (float*)   (ws + (size_t)44040192);//  8 MB
    _Float16* y16   = (_Float16*)(ws + (size_t)52428800);//  4 MB

    // weight conversion to f16
    {
        const int n1 = D_MODEL * N3;       // 6291456
        f32_to_f16_kernel<<<(n1 + 255) / 256, 256, 0, stream>>>(in_w, w1_16, n1);
        const int n2 = INNER * D_MODEL;    // 2097152
        f32_to_f16_kernel<<<(n2 + 255) / 256, 256, 0, stream>>>(out_w, w2_16, n2);
    }

    // layernorm -> h16
    layernorm_f16_kernel<<<L_SEQ, 256, 0, stream>>>(x, gamma, beta, h16);

    // GEMM1: uvg = h16 @ w1 + in_b
    wmma_gemm_f16_kernel<<<dim3(N3 / BN, L_SEQ / BM), 256, 0, stream>>>(
        h16, w1_16, in_b, nullptr, uvg, D_MODEL, D_MODEL, N3, N3);

    // depthwise conv3 + silu on u-slice of uvg
    conv_silu_kernel<<<(L_SEQ * INNER + 255) / 256, 256, 0, stream>>>(
        uvg, conv_w, conv_b, ucv);

    // sequential SSM scan + gating -> y16
    ssm_scan_kernel<<<(4 * INNER) / 256, 256, 0, stream>>>(
        ucv, uvg, A_log, B_p, C_p, D_p, y16);

    // GEMM2: out = y16 @ w2 + out_b + x
    wmma_gemm_f16_kernel<<<dim3(D_MODEL / BN, L_SEQ / BM), 256, 0, stream>>>(
        y16, w2_16, out_b, x, out, INNER, INNER, D_MODEL, D_MODEL);
}
